// TinyStoriesForCausalLM_65695819759828
// MI455X (gfx1250) — compile-verified, matrix operand order checked by reading, not run on hardware
//
#include <hip/hip_runtime.h>
#include <stdint.h>

// ---------------- model constants ----------------
#define B_   2
#define S_   1024
#define H_   1024
#define NH_  16
#define DH_  64
#define DFF_ 4096
#define L_   4
#define V_   50257
#define WIN_ 256
#define M_   (B_ * S_)   // 2048 rows of activations

typedef __attribute__((ext_vector_type(16))) __bf16 v16bf;
typedef __attribute__((ext_vector_type(8)))  float  v8f;

struct alignas(16) U128 { unsigned int w[4]; };
struct alignas(8)  U64  { unsigned int w[2]; };

__device__ __forceinline__ unsigned short f2bf(float f) {
  unsigned int u = __float_as_uint(f);
  u += 0x7FFFu + ((u >> 16) & 1u);          // round-to-nearest-even
  return (unsigned short)(u >> 16);
}

// LDS byte offset of a generic pointer to __shared__ memory:
// flat LDS addresses carry the LDS offset in addr[31:0] (aperture mapping).
__device__ __forceinline__ unsigned lds_off(const void* p) {
  return (unsigned)(size_t)p;
}

// Async DMA: 16 bytes per active lane, global -> LDS (ASYNCcnt-tracked).
__device__ __forceinline__ void async_copy_b128(unsigned lds_byte_off, const void* gptr) {
  asm volatile("global_load_async_to_lds_b128 %0, %1, off"
               :: "v"(lds_byte_off), "v"((unsigned long long)(size_t)gptr)
               : "memory");
}
__device__ __forceinline__ void wait_async0() {
  asm volatile("s_wait_asynccnt 0x0" ::: "memory");
}

// ---------------- embedding ----------------
__global__ __launch_bounds__(256) void embed_kernel(const int* __restrict__ ids,
                                                    const float* __restrict__ wemb,
                                                    const float* __restrict__ pemb,
                                                    float* __restrict__ x) {
  int row = blockIdx.x;              // 0..M-1  (b*S + s)
  int s   = row & (S_ - 1);
  int id  = ids[row];
  int c   = threadIdx.x * 4;
  const float4 w = *(const float4*)(wemb + (size_t)id * H_ + c);
  const float4 p = *(const float4*)(pemb + (size_t)s  * H_ + c);
  float4 o; o.x = w.x + p.x; o.y = w.y + p.y; o.z = w.z + p.z; o.w = w.w + p.w;
  *(float4*)(x + (size_t)row * H_ + c) = o;
}

// ---------------- layernorm: fp32 in -> bf16 out ----------------
__global__ __launch_bounds__(256) void ln_kernel(const float* __restrict__ x,
                                                 const float* __restrict__ w,
                                                 const float* __restrict__ b,
                                                 unsigned short* __restrict__ out) {
  int row = blockIdx.x;
  int tid = threadIdx.x;
  int c   = tid * 4;
  const float4 v = *(const float4*)(x + (size_t)row * H_ + c);
  float sum = v.x + v.y + v.z + v.w;
  float sq  = v.x*v.x + v.y*v.y + v.z*v.z + v.w*v.w;
  for (int d = 1; d < 32; d <<= 1) {
    sum += __shfl_xor(sum, d, 32);
    sq  += __shfl_xor(sq,  d, 32);
  }
  __shared__ float s_sum[8], s_sq[8];
  int wv = tid >> 5;
  if ((tid & 31) == 0) { s_sum[wv] = sum; s_sq[wv] = sq; }
  __syncthreads();
  float ts = 0.f, tq = 0.f;
  #pragma unroll
  for (int i = 0; i < 8; i++) { ts += s_sum[i]; tq += s_sq[i]; }
  float mu   = ts * (1.0f / H_);
  float var  = tq * (1.0f / H_) - mu * mu;
  float rstd = rsqrtf(var + 1e-5f);
  const float4 wt = *(const float4*)(w + c);
  const float4 bt = *(const float4*)(b + c);
  unsigned short o0 = f2bf((v.x - mu) * rstd * wt.x + bt.x);
  unsigned short o1 = f2bf((v.y - mu) * rstd * wt.y + bt.y);
  unsigned short o2 = f2bf((v.z - mu) * rstd * wt.z + bt.z);
  unsigned short o3 = f2bf((v.w - mu) * rstd * wt.w + bt.w);
  U64 pk; pk.w[0] = (unsigned)o0 | ((unsigned)o1 << 16);
          pk.w[1] = (unsigned)o2 | ((unsigned)o3 << 16);
  *(U64*)(out + (size_t)row * H_ + c) = pk;
}

// ---------------- tiled WMMA GEMM ----------------
// C[M,N] = A(bf16)[M,K] @ B(fp32)[K,N] (+bias)(+gelu | +residual)
// block = 256 thr (8 waves), tile 128x128, K-step 32, wave -> 64x32
// A tile staged by async DMA (global_load_async_to_lds_b128);
// B tile converted fp32->bf16 + transposed through VALU (DMA can't transform).
#define EPI_NONE 0
#define EPI_GELU 1
#define EPI_RES  2

template<int EPI, bool OUTBF>
__global__ __launch_bounds__(256) void gemm_kernel(const unsigned short* __restrict__ A,
                                                   const float* __restrict__ Bw,
                                                   const float* __restrict__ bias,
                                                   const float* __restrict__ R,
                                                   void* __restrict__ Cout,
                                                   int M, int N, int K) {
  __shared__ unsigned short lA[128 * 40];   // [row][k]  (pad 40 keeps 16B align)
  __shared__ unsigned short lB[128 * 40];   // transposed: [n][k]
  const int tid  = threadIdx.x;
  const int lane = tid & 31;
  const int wv   = tid >> 5;
  const int wm   = wv & 1;        // 2 waves over M (64 each)
  const int wn   = wv >> 1;       // 4 waves over N (32 each)
  const int m0   = blockIdx.y * 128;
  const int n0   = blockIdx.x * 128;
  const int half = (lane < 16) ? 0 : 8;     // A/B-frag K-base per lane-half
  const int nlan = lane & 15;
  const unsigned lA_base = lds_off(&lA[0]);

  v8f acc[4][2];
  #pragma unroll
  for (int i = 0; i < 4; i++)
    #pragma unroll
    for (int j = 0; j < 2; j++)
      #pragma unroll
      for (int e = 0; e < 8; e++) acc[i][j][e] = 0.f;

  for (int k0 = 0; k0 < K; k0 += 32) {
    // ---- stage A tile (bf16, 128x32) via async DMA: 512 x 16B chunks ----
    #pragma unroll
    for (int it = 0; it < 2; it++) {
      int cch = tid + it * 256;         // 0..511
      int r   = cch >> 2;               // row 0..127
      int kc  = (cch & 3) * 8;          // ushort offset 0,8,16,24
      int gm  = m0 + r;
      if (gm < M)
        async_copy_b128(lA_base + (unsigned)(r * 40 + kc) * 2,
                        A + (size_t)gm * K + k0 + kc);
    }
    // ---- stage B tile (fp32 -> bf16, transpose to [n][k]) ----
    #pragma unroll
    for (int it = 0; it < 16; it++) {
      int e = tid + it * 256;           // 0..4095
      int n = e & 127;
      int k = e >> 7;                   // 0..31
      int gn = n0 + n;
      float vv = (gn < N) ? Bw[(size_t)(k0 + k) * N + gn] : 0.f;
      lB[n * 40 + k] = f2bf(vv);
    }
    if (k0 + 32 < K)                    // stream-ahead the weight tile
      __builtin_prefetch(Bw + (size_t)(k0 + 32) * N + n0 + (tid & 127), 0, 1);
    wait_async0();                      // our DMA writes visible before barrier
    __syncthreads();

    // ---- fragments + 8 WMMA ----
    v16bf afr[4], bfr[2];
    #pragma unroll
    for (int mf = 0; mf < 4; mf++) {
      const unsigned short* p = &lA[(wm * 64 + mf * 16 + nlan) * 40 + half];
      ((U128*)&afr[mf])[0] = *(const U128*)p;
      ((U128*)&afr[mf])[1] = *(const U128*)(p + 16);
    }
    #pragma unroll
    for (int nf = 0; nf < 2; nf++) {
      const unsigned short* p = &lB[(wn * 32 + nf * 16 + nlan) * 40 + half];
      ((U128*)&bfr[nf])[0] = *(const U128*)p;
      ((U128*)&bfr[nf])[1] = *(const U128*)(p + 16);
    }
    #pragma unroll
    for (int mf = 0; mf < 4; mf++)
      #pragma unroll
      for (int nf = 0; nf < 2; nf++)
        acc[mf][nf] = __builtin_amdgcn_wmma_f32_16x16x32_bf16(
            false, afr[mf], false, bfr[nf], (short)0, acc[mf][nf], false, false);
    __syncthreads();                    // frag reads done before re-staging
  }

  // ---- epilogue ----
  const int rowoff = (lane < 16) ? 0 : 8;   // C layout: lanes>=16 hold M=r+8
  #pragma unroll
  for (int mf = 0; mf < 4; mf++)
    #pragma unroll
    for (int nf = 0; nf < 2; nf++)
      #pragma unroll
      for (int r = 0; r < 8; r++) {
        int gm = m0 + wm * 64 + mf * 16 + r + rowoff;
        int gn = n0 + wn * 32 + nf * 16 + nlan;
        if (gm < M && gn < N) {
          float vv = acc[mf][nf][r];
          if (bias) vv += bias[gn];
          if (EPI == EPI_GELU) vv = 0.5f * vv * (1.f + erff(vv * 0.70710678118f));
          if (EPI == EPI_RES)  vv += R[(size_t)gm * N + gn];
          if (OUTBF) ((unsigned short*)Cout)[(size_t)gm * N + gn] = f2bf(vv);
          else       ((float*)Cout)[(size_t)gm * N + gn] = vv;
        }
      }
}

// ---------------- windowed flash attention ----------------
// one wave (32 thr) per (b, head, 16-query tile); keys tiled by 32.
// V tile staged by async DMA, overlapped with Q*K^T and softmax math.
__global__ __launch_bounds__(32) void attn_kernel(const unsigned short* __restrict__ q,
                                                  const unsigned short* __restrict__ kk,
                                                  const unsigned short* __restrict__ vv,
                                                  unsigned short* __restrict__ ctx) {
  __shared__ unsigned short lP[16 * 40];    // P tile (bf16), A-layout staging
  __shared__ unsigned short lV[32 * 64];    // V tile [key][dh]
  const int lane = threadIdx.x;
  const int i0   = blockIdx.x * 16;
  const int hh   = blockIdx.y;
  const int bb   = blockIdx.z;
  const int halfk  = (lane < 16) ? 0 : 8;
  const int rowoff = (lane < 16) ? 0 : 8;
  const int nlan   = lane & 15;
  const unsigned lV_base = lds_off(&lV[0]);

  // Q A-frags (rows i0..i0+15, dh chunks 0-31 / 32-63)
  v16bf qa[2];
  {
    const unsigned short* qp = q + ((size_t)(bb * S_ + i0 + nlan)) * H_ + hh * DH_;
    #pragma unroll
    for (int c = 0; c < 2; c++) {
      int kb = c * 32 + halfk;
      ((U128*)&qa[c])[0] = *(const U128*)(qp + kb);
      ((U128*)&qa[c])[1] = *(const U128*)(qp + kb + 16);
    }
  }

  v8f o[4];
  float mrow[8], lrow[8], alpha[8];
  #pragma unroll
  for (int nf = 0; nf < 4; nf++)
    #pragma unroll
    for (int e = 0; e < 8; e++) o[nf][e] = 0.f;
  #pragma unroll
  for (int r = 0; r < 8; r++) { mrow[r] = -1e30f; lrow[r] = 0.f; }

  const float scale = 0.125f;               // 1/sqrt(64)
  int jbeg = i0 - (WIN_ - 1); if (jbeg < 0) jbeg = 0;
  jbeg &= ~31;
  const int jend = i0 + 15;

  for (int j0 = jbeg; j0 <= jend; j0 += 32) {
    // kick off V tile DMA (row per lane, 128B each) -- overlapped with QK^T
    {
      int jr = j0 + lane; if (jr > S_ - 1) jr = S_ - 1;
      const unsigned short* vp = vv + ((size_t)(bb * S_ + jr)) * H_ + hh * DH_;
      unsigned ldst = lV_base + (unsigned)lane * 128;
      #pragma unroll
      for (int t = 0; t < 8; t++)
        async_copy_b128(ldst + t * 16, vp + t * 8);
    }
    // scores: two 16-key chunks, each = 2 WMMA over dh
    v8f sc[2];
    #pragma unroll
    for (int nc = 0; nc < 2; nc++) {
      int j  = j0 + nc * 16 + nlan;
      int jc = (j > S_ - 1) ? (S_ - 1) : j;
      const unsigned short* kp = kk + ((size_t)(bb * S_ + jc)) * H_ + hh * DH_;
      v8f c;
      #pragma unroll
      for (int e = 0; e < 8; e++) c[e] = 0.f;
      #pragma unroll
      for (int cc = 0; cc < 2; cc++) {
        v16bf kf;
        int kb = cc * 32 + halfk;
        ((U128*)&kf)[0] = *(const U128*)(kp + kb);
        ((U128*)&kf)[1] = *(const U128*)(kp + kb + 16);
        c = __builtin_amdgcn_wmma_f32_16x16x32_bf16(false, qa[cc], false, kf,
                                                    (short)0, c, false, false);
      }
      #pragma unroll
      for (int r = 0; r < 8; r++) {
        int i = i0 + r + rowoff;
        bool ok = (j <= i) && (j > i - WIN_);
        c[r] = ok ? c[r] * scale : -1e30f;
      }
      sc[nc] = c;
    }
    // online softmax (row stats per vgpr slot, reduced over 16-lane halves)
    #pragma unroll
    for (int r = 0; r < 8; r++) {
      float t = fmaxf(sc[0][r], sc[1][r]);
      t = fmaxf(t, __shfl_xor(t, 1, 32));
      t = fmaxf(t, __shfl_xor(t, 2, 32));
      t = fmaxf(t, __shfl_xor(t, 4, 32));
      t = fmaxf(t, __shfl_xor(t, 8, 32));
      float mnew = fmaxf(mrow[r], t);
      alpha[r] = __expf(mrow[r] - mnew);
      float p0 = (sc[0][r] > -1e29f) ? __expf(sc[0][r] - mnew) : 0.f;
      float p1 = (sc[1][r] > -1e29f) ? __expf(sc[1][r] - mnew) : 0.f;
      float s = p0 + p1;
      s += __shfl_xor(s, 1, 32); s += __shfl_xor(s, 2, 32);
      s += __shfl_xor(s, 4, 32); s += __shfl_xor(s, 8, 32);
      lrow[r] = lrow[r] * alpha[r] + s;
      mrow[r] = mnew;
      int m = r + rowoff;
      lP[m * 40 + nlan]      = f2bf(p0);   // C-layout -> A-layout via LDS
      lP[m * 40 + 16 + nlan] = f2bf(p1);
    }
    #pragma unroll
    for (int nf = 0; nf < 4; nf++)
      #pragma unroll
      for (int r = 0; r < 8; r++) o[nf][r] *= alpha[r];

    asm volatile("s_wait_dscnt 0x0" ::: "memory");   // lP visible
    wait_async0();                                   // lV DMA complete

    // P A-frag, V B-frags, 4 WMMA (K = 32 keys)
    v16bf pa;
    {
      const unsigned short* pp = &lP[nlan * 40 + halfk];
      ((U128*)&pa)[0] = *(const U128*)pp;
      ((U128*)&pa)[1] = *(const U128*)(pp + 16);
    }
    #pragma unroll
    for (int nf = 0; nf < 4; nf++) {
      v16bf vb;
      unsigned short* vu = (unsigned short*)&vb;
      int d = nf * 16 + nlan;
      #pragma unroll
      for (int t = 0; t < 8; t++) {
        vu[t]     = lV[(halfk + t) * 64 + d];
        vu[t + 8] = lV[(halfk + 16 + t) * 64 + d];
      }
      o[nf] = __builtin_amdgcn_wmma_f32_16x16x32_bf16(false, pa, false, vb,
                                                      (short)0, o[nf], false, false);
    }
  }

  // normalize and write ctx (bf16)
  #pragma unroll
  for (int r = 0; r < 8; r++) {
    float inv = 1.f / lrow[r];
    int i = i0 + r + rowoff;
    size_t base = ((size_t)(bb * S_ + i)) * H_ + hh * DH_;
    #pragma unroll
    for (int nf = 0; nf < 4; nf++)
      ctx[base + nf * 16 + nlan] = f2bf(o[nf][r] * inv);
  }
}

// ---------------- host orchestration ----------------
extern "C" void kernel_launch(void* const* d_in, const int* in_sizes, int n_in,
                              void* d_out, int out_size, void* d_ws, size_t ws_size,
                              hipStream_t stream) {
  (void)in_sizes; (void)n_in; (void)out_size; (void)ws_size;
  const int*   ids  = (const int*)  d_in[0];
  const float* wemb = (const float*)d_in[1];
  const float* pemb = (const float*)d_in[2];
  const float* ln1w = (const float*)d_in[3];
  const float* ln1b = (const float*)d_in[4];
  const float* Wq   = (const float*)d_in[5];
  const float* bq   = (const float*)d_in[6];
  const float* Wk   = (const float*)d_in[7];
  const float* bk   = (const float*)d_in[8];
  const float* Wv   = (const float*)d_in[9];
  const float* bv   = (const float*)d_in[10];
  const float* Wo   = (const float*)d_in[11];
  const float* bo   = (const float*)d_in[12];
  const float* ln2w = (const float*)d_in[13];
  const float* ln2b = (const float*)d_in[14];
  const float* W1   = (const float*)d_in[15];
  const float* b1   = (const float*)d_in[16];
  const float* W2   = (const float*)d_in[17];
  const float* b2   = (const float*)d_in[18];
  const float* lnfw = (const float*)d_in[19];
  const float* lnfb = (const float*)d_in[20];
  const float* lmh  = (const float*)d_in[21];
  float* logits = (float*)d_out;

  // workspace layout (~44 MB)
  char* ws = (char*)d_ws;
  float*          x   = (float*)ws;          ws += (size_t)M_ * H_   * 4;
  unsigned short* h   = (unsigned short*)ws; ws += (size_t)M_ * H_   * 2;
  unsigned short* qb  = (unsigned short*)ws; ws += (size_t)M_ * H_   * 2;
  unsigned short* kbuf= (unsigned short*)ws; ws += (size_t)M_ * H_   * 2;
  unsigned short* vbuf= (unsigned short*)ws; ws += (size_t)M_ * H_   * 2;
  unsigned short* ctx = (unsigned short*)ws; ws += (size_t)M_ * H_   * 2;
  unsigned short* ffn = (unsigned short*)ws; ws += (size_t)M_ * DFF_ * 2;

  embed_kernel<<<M_, 256, 0, stream>>>(ids, wemb, pemb, x);

  dim3 gH(H_ / 128, M_ / 128);                // 8 x 16
  dim3 gF(DFF_ / 128, M_ / 128);              // 32 x 16
  dim3 gV((V_ + 127) / 128, M_ / 128);        // 393 x 16
  dim3 gA(S_ / 16, NH_, B_);                  // 64 x 16 x 2

  for (int l = 0; l < L_; l++) {
    ln_kernel<<<M_, 256, 0, stream>>>(x, ln1w + l * H_, ln1b + l * H_, h);
    gemm_kernel<EPI_NONE, true><<<gH, 256, 0, stream>>>(h, Wq + (size_t)l * H_ * H_, bq + l * H_, nullptr, qb,   M_, H_, H_);
    gemm_kernel<EPI_NONE, true><<<gH, 256, 0, stream>>>(h, Wk + (size_t)l * H_ * H_, bk + l * H_, nullptr, kbuf, M_, H_, H_);
    gemm_kernel<EPI_NONE, true><<<gH, 256, 0, stream>>>(h, Wv + (size_t)l * H_ * H_, bv + l * H_, nullptr, vbuf, M_, H_, H_);
    attn_kernel<<<gA, 32, 0, stream>>>(qb, kbuf, vbuf, ctx);
    gemm_kernel<EPI_RES, false><<<gH, 256, 0, stream>>>(ctx, Wo + (size_t)l * H_ * H_, bo + l * H_, x, x, M_, H_, H_);
    ln_kernel<<<M_, 256, 0, stream>>>(x, ln2w + l * H_, ln2b + l * H_, h);
    gemm_kernel<EPI_GELU, true><<<gF, 256, 0, stream>>>(h, W1 + (size_t)l * H_ * DFF_, b1 + l * DFF_, nullptr, ffn, M_, DFF_, H_);
    gemm_kernel<EPI_RES, false><<<gH, 256, 0, stream>>>(ffn, W2 + (size_t)l * DFF_ * H_, b2 + l * H_, x, x, M_, H_, DFF_);
  }

  ln_kernel<<<M_, 256, 0, stream>>>(x, lnfw, lnfb, h);
  gemm_kernel<EPI_NONE, false><<<gV, 256, 0, stream>>>(h, lmh, nullptr, nullptr, logits, M_, V_, H_);
}
